// V2XTEncoder_43001212567578
// MI455X (gfx1250) — compile-verified, matrix operand order checked by reading, not run on hardware
//
#include <hip/hip_runtime.h>
#include <cstdint>
#include <cstddef>

typedef _Float16 half_t;
typedef __attribute__((ext_vector_type(16))) _Float16 v16h;
typedef __attribute__((ext_vector_type(8)))  _Float16 v8h;
typedef __attribute__((ext_vector_type(8)))  float    v8f;

// exact param types for the gfx1250 async global->LDS builtins (from clang diag)
typedef int v4i_t __attribute__((vector_size(4 * sizeof(int))));
typedef __attribute__((address_space(1))) v4i_t* gvec_ptr;   // global (__device__) int4*
typedef __attribute__((address_space(3))) v4i_t* lvec_ptr;   // LDS (__shared__) int4*

#define DIMC   256
#define NHEADS 8
#define DHEAD  32
#define LFR    5
#define HH     48
#define WW     176
#define TOKL   (HH*WW)      // 8448
#define MTOT   (LFR*TOKL)   // 42240
#define XCH    (DIMC+3)     // 259

// CDNA5 async global->LDS copy path (ASYNCcnt), if the toolchain exposes it.
#if defined(__has_builtin)
#if __has_builtin(__builtin_amdgcn_global_load_async_to_lds_b128) && \
    __has_builtin(__builtin_amdgcn_s_wait_asynccnt)
#define USE_ASYNC_LDS 1
#endif
#endif
#ifndef USE_ASYNC_LDS
#define USE_ASYNC_LDS 0
#endif

// ---------------------------------------------------------------------------
// WMMA GEMM:  C[M,N] = A[M,K](f16 row-major) * B[N,K](f16, i.e. W^T) + epilogue
// Block tile 128x128, K-step 32, 8 waves (4x2), wave tile 32x64 = 2x4 WMMA.
// ---------------------------------------------------------------------------
enum { EP_BIAS=1, EP_GELU=2, EP_RESID=4, EP_ACCUM=8, EP_HALFOUT=16 };

__global__ void __launch_bounds__(256)
wmma_gemm_kernel(const half_t* __restrict__ A, int lda,
                 const half_t* __restrict__ Bw, int ldb,
                 const float* __restrict__ bias,
                 float* __restrict__ C, half_t* __restrict__ Ch, int ldc,
                 const float* __restrict__ R, int ldr,
                 int K,
                 const int* __restrict__ sel, int rows_per_sel,
                 int selB_stride, int selBias_stride,
                 int flags)
{
    constexpr int BM = 128, BN = 128, BK = 32, LS = 40;  // padded rows: 80B, 16B aligned
    __shared__ half_t As[BM*LS];
    __shared__ half_t Bs[BN*LS];

    const int tid  = threadIdx.x;
    const int lane = tid & 31;
    const int wid  = tid >> 5;
    const int wm   = wid >> 1;          // 0..3 -> 32-row stripes
    const int wn   = wid & 1;           // 0..1 -> 64-col stripes
    const int bm0  = blockIdx.x * BM;
    const int bn0  = blockIdx.y * BN;

    const half_t* Ab = A;
    const half_t* Bb = Bw;
    const float*  bp = bias;
    if (sel) {                           // uniform across the workgroup
        int s = sel[bm0 / rows_per_sel];
        Bb += (size_t)s * selB_stride;
        if (bp) bp += (size_t)s * selBias_stride;
    }

    v8f acc[2][4];
    #pragma unroll
    for (int i = 0; i < 2; ++i)
        #pragma unroll
        for (int j = 0; j < 4; ++j)
            #pragma unroll
            for (int e = 0; e < 8; ++e) acc[i][j][e] = 0.f;

    // fragment lane mapping (16-bit A 16x32 layout, ISA 7.12.2)
    const int lrow = lane & 15;
    const int lko  = (lane >> 4) * 8;   // K chunks {0..7,16..23} vs {8..15,24..31}

    // cooperative tile-load mapping: 256 threads, v8h each; rows r and r+64
    const int ldrw = tid >> 2;          // 0..63
    const int ldk  = (tid & 3) * 8;

    for (int k0 = 0; k0 < K; k0 += BK) {
        const half_t* ga0 = Ab + (size_t)(bm0+ldrw   )*lda + k0 + ldk;
        const half_t* ga1 = Ab + (size_t)(bm0+ldrw+64)*lda + k0 + ldk;
        const half_t* gb0 = Bb + (size_t)(bn0+ldrw   )*ldb + k0 + ldk;
        const half_t* gb1 = Bb + (size_t)(bn0+ldrw+64)*ldb + k0 + ldk;
#if USE_ASYNC_LDS
        __builtin_amdgcn_global_load_async_to_lds_b128(
            (gvec_ptr)ga0, (lvec_ptr)&As[ldrw*LS + ldk], 0, 0);
        __builtin_amdgcn_global_load_async_to_lds_b128(
            (gvec_ptr)ga1, (lvec_ptr)&As[(ldrw+64)*LS + ldk], 0, 0);
        __builtin_amdgcn_global_load_async_to_lds_b128(
            (gvec_ptr)gb0, (lvec_ptr)&Bs[ldrw*LS + ldk], 0, 0);
        __builtin_amdgcn_global_load_async_to_lds_b128(
            (gvec_ptr)gb1, (lvec_ptr)&Bs[(ldrw+64)*LS + ldk], 0, 0);
        __builtin_amdgcn_s_wait_asynccnt(0);
#else
        *(v8h*)&As[ldrw*LS + ldk]      = *(const v8h*)ga0;
        *(v8h*)&As[(ldrw+64)*LS + ldk] = *(const v8h*)ga1;
        *(v8h*)&Bs[ldrw*LS + ldk]      = *(const v8h*)gb0;
        *(v8h*)&Bs[(ldrw+64)*LS + ldk] = *(const v8h*)gb1;
#endif
        __syncthreads();

        if (k0 + BK < K)   // global_prefetch_b8 of next A slice
            __builtin_prefetch(Ab + (size_t)(bm0+ldrw)*lda + k0 + BK + ldk, 0, 3);

        v16h afr[2], bfr[4];
        #pragma unroll
        for (int i = 0; i < 2; ++i) {
            const half_t* pa = &As[(wm*32 + i*16 + lrow)*LS + lko];
            v8h lo = *(const v8h*)pa;
            v8h hi = *(const v8h*)(pa + 16);
            #pragma unroll
            for (int e = 0; e < 8; ++e) { afr[i][e] = lo[e]; afr[i][8+e] = hi[e]; }
        }
        #pragma unroll
        for (int j = 0; j < 4; ++j) {
            const half_t* pb = &Bs[(wn*64 + j*16 + lrow)*LS + lko];
            v8h lo = *(const v8h*)pb;
            v8h hi = *(const v8h*)(pb + 16);
            #pragma unroll
            for (int e = 0; e < 8; ++e) { bfr[j][e] = lo[e]; bfr[j][8+e] = hi[e]; }
        }

        #pragma unroll
        for (int i = 0; i < 2; ++i)
            #pragma unroll
            for (int j = 0; j < 4; ++j)
                acc[i][j] = __builtin_amdgcn_wmma_f32_16x16x32_f16(
                    false, afr[i], false, bfr[j], (short)0, acc[i][j], false, false);
        __syncthreads();
    }

    // C/D layout: VGPR r -> M = r + (lane>=16 ? 8:0), N = lane&15
    const int cmb = (lane >> 4) * 8;
    const int cn  = lane & 15;
    #pragma unroll
    for (int i = 0; i < 2; ++i)
        #pragma unroll
        for (int j = 0; j < 4; ++j) {
            const int col = bn0 + wn*64 + j*16 + cn;
            #pragma unroll
            for (int rr = 0; rr < 8; ++rr) {
                const int row = bm0 + wm*32 + i*16 + cmb + rr;
                float v = acc[i][j][rr];
                if (flags & EP_BIAS)  v += bp[col];
                if (flags & EP_GELU)  v = 0.5f * v * (1.f + erff(v * 0.70710678f));
                if (flags & EP_ACCUM) v += C[(size_t)row*ldc + col];
                if (flags & EP_RESID) v += R[(size_t)row*ldr + col];
                if (flags & EP_HALFOUT) Ch[(size_t)row*ldc + col] = (half_t)v;
                else                    C [(size_t)row*ldc + col] = v;
            }
        }
}

// ---------------------------------------------------------------------------
// Weight convert: fp32 [b][K][N] -> f16 [b][N][K] (transpose for B-fragments)
// ---------------------------------------------------------------------------
__global__ void convw_kernel(const float* __restrict__ src, half_t* __restrict__ dst,
                             int K, int N, int total)
{
    int id = blockIdx.x * blockDim.x + threadIdx.x;
    if (id >= total) return;
    int b = id / (K*N);
    int r = id % (K*N);
    int n = r / K, k = r % K;
    dst[(size_t)b*K*N + (size_t)n*K + k] = (half_t)src[(size_t)b*K*N + (size_t)k*N + n];
}

// ---------------------------------------------------------------------------
// LayerNorm over 256 dims, one wave32 per token, output f16 GEMM-A buffer.
// ---------------------------------------------------------------------------
__global__ void __launch_bounds__(256)
ln_kernel(const float* __restrict__ x, const float* __restrict__ g,
          const float* __restrict__ b, half_t* __restrict__ out)
{
    const int row  = blockIdx.x * 8 + (threadIdx.x >> 5);
    const int lane = threadIdx.x & 31;
    const float* r = x + (size_t)row * DIMC;
    float v[8];
    #pragma unroll
    for (int i = 0; i < 4; ++i) v[i]   = r[lane*4 + i];
    #pragma unroll
    for (int i = 0; i < 4; ++i) v[4+i] = r[128 + lane*4 + i];
    float s = 0.f, s2 = 0.f;
    #pragma unroll
    for (int i = 0; i < 8; ++i) { s += v[i]; s2 += v[i]*v[i]; }
    #pragma unroll
    for (int m = 16; m >= 1; m >>= 1) { s += __shfl_xor(s, m, 32); s2 += __shfl_xor(s2, m, 32); }
    const float mu = s * (1.f/256.f);
    const float var = s2 * (1.f/256.f) - mu*mu;
    const float rs = rsqrtf(var + 1e-5f);
    half_t* o = out + (size_t)row * DIMC;
    #pragma unroll
    for (int i = 0; i < 4; ++i) { int c = lane*4+i;     o[c] = (half_t)((v[i]  -mu)*rs*g[c] + b[c]); }
    #pragma unroll
    for (int i = 0; i < 4; ++i) { int c = 128+lane*4+i; o[c] = (half_t)((v[4+i]-mu)*rs*g[c] + b[c]); }
}

// ---------------------------------------------------------------------------
// Meta extraction: dt[l] and type[l] from x[..,0,0,257/258]
// ---------------------------------------------------------------------------
__global__ void extract_meta_kernel(const float* __restrict__ x, int* __restrict__ meta)
{
    int l = threadIdx.x;
    if (l < LFR) {
        size_t base = (size_t)l * TOKL * XCH;
        meta[l]     = (int)x[base + DIMC + 1];   // dt
        meta[8 + l] = (int)x[base + DIMC + 2];   // type
    }
}

// ---------------------------------------------------------------------------
// RTE sinusoid embedding: temb[l] = tab[dt*2] @ rte_w + rte_b
// ---------------------------------------------------------------------------
__global__ void rte_kernel(const int* __restrict__ meta,
                           const float* __restrict__ rte_w,
                           const float* __restrict__ rte_b,
                           float* __restrict__ temb)
{
    int id = blockIdx.x * blockDim.x + threadIdx.x;
    if (id >= LFR * DIMC) return;
    const int l = id >> 8, n = id & 255;
    const float pos = (float)(meta[l] * 2);
    const float c0 = -logf(10000.f) / (float)DIMC;
    float acc = 0.f;
    for (int k = 0; k < DIMC/2; ++k) {
        float ang = pos * expf((float)(2*k) * c0);
        acc += sinf(ang) * 0.0625f * rte_w[(2*k  )*DIMC + n];
        acc += cosf(ang) * 0.0625f * rte_w[(2*k+1)*DIMC + n];
    }
    temb[id] = acc + rte_b[n];
}

__global__ void build_feat_kernel(const float* __restrict__ x,
                                  const float* __restrict__ temb,
                                  float* __restrict__ feat)
{
    int id = blockIdx.x * blockDim.x + threadIdx.x;   // MTOT*256
    int t = id >> 8, c = id & 255;
    int l = t / TOKL;
    feat[id] = x[(size_t)t * XCH + c] + temb[l*DIMC + c];
}

// ---------------------------------------------------------------------------
// STTF: per-frame affine theta then bilinear grid-sample.
// ---------------------------------------------------------------------------
__device__ inline void mat3mul(float* o, const float* a, const float* b) {
    for (int i = 0; i < 3; ++i)
        for (int j = 0; j < 3; ++j)
            o[i*3+j] = a[i*3]*b[j] + a[i*3+1]*b[3+j] + a[i*3+2]*b[6+j];
}
__device__ inline void mat3inv(float* o, const float* m) {
    float a=m[0],b=m[1],c=m[2],d=m[3],e=m[4],f=m[5],g=m[6],h=m[7],i=m[8];
    float A = e*i - f*h, B = -(d*i - f*g), Cc = d*h - e*g;
    float inv = 1.f / (a*A + b*B + c*Cc);
    o[0]=A*inv;   o[1]=-(b*i-c*h)*inv; o[2]=(b*f-c*e)*inv;
    o[3]=B*inv;   o[4]=(a*i-c*g)*inv;  o[5]=-(a*f-c*d)*inv;
    o[6]=Cc*inv;  o[7]=-(a*h-b*g)*inv; o[8]=(a*e-b*d)*inv;
}

__global__ void theta_kernel(const float* __restrict__ scm, float* __restrict__ theta)
{
    int l = threadIdx.x;
    if (l >= LFR) return;
    const float* S = scm + l*16;
    const float inv_vd = 1.0f / (0.4f * 4.0f);
    float R00=S[0], R01=S[1], tx=S[3]*inv_vd;
    float R10=S[4], R11=S[5], ty=S[7]*inv_vd;
    const float cx = WW*0.5f, cy = HH*0.5f;
    float t0 = cx - (R00*cx + R01*cy) + tx;
    float t1 = cy - (R10*cx + R11*cy) + ty;
    float T [9] = {R00,R01,t0,  R10,R11,t1,  0,0,1};
    float Nm[9] = {2.f/(WW-1),0,-1,  0,2.f/(HH-1),-1,  0,0,1};
    float Ni[9] = {(WW-1)*0.5f,0,(WW-1)*0.5f,  0,(HH-1)*0.5f,(HH-1)*0.5f,  0,0,1};
    float A[9], D[9], Di[9];
    mat3mul(A, Nm, T);
    mat3mul(D, A, Ni);
    mat3inv(Di, D);
    for (int k = 0; k < 6; ++k) theta[l*6 + k] = Di[k];
}

__global__ void sttf_kernel(const float* __restrict__ feat0,
                            const float* __restrict__ theta,
                            float* __restrict__ feat)
{
    int id = blockIdx.x * blockDim.x + threadIdx.x;   // MTOT*32 (8 channels each)
    int token = id >> 5;
    int ch = (id & 31) * 8;
    int l = token / TOKL, rem = token % TOKL;
    int y = rem / WW, xc = rem % WW;
    const float* th = theta + l*6;
    float xs = (2.f*xc + 1.f)/WW - 1.f;
    float ys = (2.f*y  + 1.f)/HH - 1.f;
    float g0 = th[0]*xs + th[1]*ys + th[2];
    float g1 = th[3]*xs + th[4]*ys + th[5];
    float gx = ((g0 + 1.f)*WW - 1.f)*0.5f;
    float gy = ((g1 + 1.f)*HH - 1.f)*0.5f;
    float x0 = floorf(gx), y0 = floorf(gy);
    float wx = gx - x0, wy = gy - y0;
    int xi0 = (int)x0, yi0 = (int)y0;
    float w [4] = {(1-wx)*(1-wy), wx*(1-wy), (1-wx)*wy, wx*wy};
    int   xt[4] = {xi0, xi0+1, xi0, xi0+1};
    int   yt[4] = {yi0, yi0, yi0+1, yi0+1};
    float o[8] = {0,0,0,0,0,0,0,0};
    #pragma unroll
    for (int t = 0; t < 4; ++t) {
        if (xt[t] < 0 || xt[t] >= WW || yt[t] < 0 || yt[t] >= HH) continue;
        const float* src = feat0 + (((size_t)l*HH + yt[t])*WW + xt[t])*DIMC + ch;
        float wt = w[t];
        #pragma unroll
        for (int c = 0; c < 8; ++c) o[c] += wt * src[c];
    }
    float* dst = feat + (size_t)token*DIMC + ch;
    #pragma unroll
    for (int c = 0; c < 8; ++c) dst[c] = o[c];
}

// ---------------------------------------------------------------------------
// HGT relation attention: thread per (position, head, target-frame i).
// att_ij = q_i . (W_att[e_ij,m] k_j) / sqrt(32); out_i = sum_j a_ij v_j^T W_msg
// ---------------------------------------------------------------------------
__global__ void __launch_bounds__(256)
hgt_attn_kernel(const float* __restrict__ qkv,
                const float* __restrict__ rel_att,
                const float* __restrict__ rel_msg,
                const float* __restrict__ mask,
                const int* __restrict__ meta,
                half_t* __restrict__ out)
{
    int id = blockIdx.x * 256 + threadIdx.x;   // TOKL*NHEADS*LFR
    int p = id / (NHEADS * LFR);
    int r = id % (NHEADS * LFR);
    int m = r / LFR;
    int i = r % LFR;
    const int* tp = meta + 8;
    const int ti = i * TOKL + p;

    float q[DHEAD];
    const float* qp = qkv + (size_t)ti*768 + m*DHEAD;
    #pragma unroll
    for (int c = 0; c < DHEAD; ++c) q[c] = qp[c];

    float att[LFR];
    float mx = -1e30f;
    #pragma unroll
    for (int j = 0; j < LFR; ++j) {
        if (mask[j] == 0.f) { att[j] = -1e30f; continue; }
        const float* kp = qkv + (size_t)(j*TOKL + p)*768 + 256 + m*DHEAD;
        float k[DHEAD];
        #pragma unroll
        for (int c = 0; c < DHEAD; ++c) k[c] = kp[c];
        const float* Wm = rel_att + ((size_t)(tp[i]*2 + tp[j])*NHEADS + m)*DHEAD*DHEAD;
        float s = 0.f;
        for (int pp = 0; pp < DHEAD; ++pp) {
            float w = 0.f;
            #pragma unroll
            for (int qq = 0; qq < DHEAD; ++qq) w += Wm[pp*DHEAD + qq] * k[qq];
            s += q[pp] * w;
        }
        att[j] = s * 0.17677669529663687f;
        mx = fmaxf(mx, att[j]);
    }
    float den = 0.f;
    #pragma unroll
    for (int j = 0; j < LFR; ++j) {
        att[j] = (mask[j] == 0.f) ? 0.f : expf(att[j] - mx);
        den += att[j];
    }
    const float inv = 1.f / den;

    float o[DHEAD] = {};
    #pragma unroll
    for (int j = 0; j < LFR; ++j) {
        float a = att[j] * inv;
        if (a == 0.f) continue;
        const float* vp = qkv + (size_t)(j*TOKL + p)*768 + 512 + m*DHEAD;
        const float* Wm = rel_msg + ((size_t)(tp[i]*2 + tp[j])*NHEADS + m)*DHEAD*DHEAD;
        for (int pp = 0; pp < DHEAD; ++pp) {
            float f = vp[pp] * a;
            #pragma unroll
            for (int c = 0; c < DHEAD; ++c) o[c] += f * Wm[pp*DHEAD + c];
        }
    }
    half_t* op = out + (size_t)ti*DIMC + m*DHEAD;
    #pragma unroll
    for (int c = 0; c < DHEAD; ++c) op[c] = (half_t)o[c];
}

// ---------------------------------------------------------------------------
// Pyramid-window attention (per level), online softmax, thread per (win,head,i)
// ---------------------------------------------------------------------------
template <int WS, int NHD, int DH>
__global__ void __launch_bounds__(256)
pw_attn_kernel(const float* __restrict__ qkv, const float* __restrict__ pos,
               half_t* __restrict__ out)
{
    constexpr int WS2 = WS*WS;
    constexpr int NW = WW/WS, NH = HH/WS;
    int id = blockIdx.x * 256 + threadIdx.x;
    int i  = id % WS2; int t1 = id / WS2;
    int m  = t1 % NHD; t1 /= NHD;
    int wx = t1 % NW;  t1 /= NW;
    int wy = t1 % NH;  int l = t1 / NH;
    const int yi = i / WS, xi = i % WS;
    const int ti = (l*HH + wy*WS + yi)*WW + wx*WS + xi;

    const float scale = rsqrtf((float)DH);
    float q[DH];
    const float* qp = qkv + (size_t)ti*768 + m*DH;
    #pragma unroll
    for (int c = 0; c < DH; ++c) q[c] = qp[c];

    float o[DH] = {};
    float mx = -1e30f, ssum = 0.f;
    for (int j = 0; j < WS2; ++j) {
        const int yj = j / WS, xj = j % WS;
        const int tj = (l*HH + wy*WS + yj)*WW + wx*WS + xj;
        const float* kp = qkv + (size_t)tj*768 + 256 + m*DH;
        float d = 0.f;
        #pragma unroll
        for (int c = 0; c < DH; ++c) d += q[c] * kp[c];
        d = d*scale + pos[(yj - yi + WS - 1)*(2*WS - 1) + (xj - xi + WS - 1)];
        float nm = fmaxf(mx, d);
        float corr = expf(mx - nm);
        float w = expf(d - nm);
        ssum = ssum*corr + w;
        const float* vp = qkv + (size_t)tj*768 + 512 + m*DH;
        #pragma unroll
        for (int c = 0; c < DH; ++c) o[c] = o[c]*corr + w*vp[c];
        mx = nm;
    }
    const float inv = 1.f / ssum;
    half_t* op = out + (size_t)ti*DIMC + m*DH;
    #pragma unroll
    for (int c = 0; c < DH; ++c) op[c] = (half_t)(o[c]*inv);
}

__global__ void add_scale_kernel(float* __restrict__ feat,
                                 const float* __restrict__ acc, float s)
{
    int id = blockIdx.x * blockDim.x + threadIdx.x;   // MTOT*256
    feat[id] += acc[id] * s;
}

// ---------------------------------------------------------------------------
// Orchestration
// ---------------------------------------------------------------------------
extern "C" void kernel_launch(void* const* d_in, const int* in_sizes, int n_in,
                              void* d_out, int out_size, void* d_ws, size_t ws_size,
                              hipStream_t stream)
{
    (void)in_sizes; (void)n_in; (void)out_size; (void)ws_size;
    const float* x     = (const float*)d_in[0];
    const float* mask  = (const float*)d_in[1];
    const float* scm   = (const float*)d_in[2];
    const float* rte_w = (const float*)d_in[3];
    const float* rte_b = (const float*)d_in[4];
    // layer input base: 5 + layer*32 (dict insertion order)
    #define LIN(ly, off) ((const float*)d_in[5 + (ly)*32 + (off)])

    // -------- workspace carve (256B aligned) --------
    char* wp = (char*)d_ws;
    auto carve = [&](size_t bytes) { void* r = wp; wp += (bytes + 255) & ~(size_t)255; return r; };
    float*  feat0 = (float*) carve((size_t)MTOT*DIMC*4);
    float*  feat  = (float*) carve((size_t)MTOT*DIMC*4);
    float*  temb  = (float*) carve((size_t)LFR*DIMC*4);
    int*    meta  = (int*)   carve(64);
    float*  theta = (float*) carve((size_t)LFR*6*4);
    half_t* Ah    = (half_t*)carve((size_t)MTOT*DIMC*2);   // LN output (GEMM A)
    half_t* Bh2   = (half_t*)carve((size_t)MTOT*DIMC*2);   // attn-out / FF hidden (GEMM A)
    float*  qkv   = (float*) carve((size_t)MTOT*768*4);
    float*  pwacc = (float*) carve((size_t)MTOT*DIMC*4);
    half_t* Wh    = (half_t*)carve((size_t)2*256*768*2);   // converted weights

    auto gemm = [&](const half_t* A, const half_t* Bw, const float* bias,
                    float* C, half_t* Ch, int ldc, const float* R, int ldr,
                    int N, const int* sel, int selBs, int selbs, int flags) {
        dim3 grid(MTOT/128, N/128);
        wmma_gemm_kernel<<<grid, 256, 0, stream>>>(A, DIMC, Bw, DIMC, bias,
                                                   C, Ch, ldc, R, ldr, DIMC,
                                                   sel, TOKL, selBs, selbs, flags);
    };
    auto convw = [&](const float* src, half_t* dst, int K, int N, int batch) {
        int total = K * N * batch;
        convw_kernel<<<(total + 255)/256, 256, 0, stream>>>(src, dst, K, N, total);
    };
    auto ln = [&](const float* src, const float* g, const float* b, half_t* dst) {
        ln_kernel<<<MTOT/8, 256, 0, stream>>>(src, g, b, dst);
    };

    // -------- preamble: meta, RTE embedding, STTF --------
    extract_meta_kernel<<<1, 32, 0, stream>>>(x, meta);
    rte_kernel<<<(LFR*DIMC + 255)/256, 256, 0, stream>>>(meta, rte_w, rte_b, temb);
    build_feat_kernel<<<(MTOT*DIMC)/256, 256, 0, stream>>>(x, temb, feat0);
    theta_kernel<<<1, 32, 0, stream>>>(scm, theta);
    sttf_kernel<<<(MTOT*32)/256, 256, 0, stream>>>(feat0, theta, feat);

    const int* typesel = meta + 8;

    for (int ly = 0; ly < 3; ++ly) {
        // ---------- HGT ----------
        ln(feat, LIN(ly,0), LIN(ly,1), Ah);
        convw(LIN(ly,2), Wh, 256, 256, 2);   // q_w (2 types)
        gemm(Ah, Wh, LIN(ly,3), qkv + 0,   nullptr, 768, nullptr, 0, 256,
             typesel, 256*256, 256, EP_BIAS);
        convw(LIN(ly,4), Wh, 256, 256, 2);   // k_w
        gemm(Ah, Wh, LIN(ly,5), qkv + 256, nullptr, 768, nullptr, 0, 256,
             typesel, 256*256, 256, EP_BIAS);
        convw(LIN(ly,6), Wh, 256, 256, 2);   // v_w
        gemm(Ah, Wh, LIN(ly,7), qkv + 512, nullptr, 768, nullptr, 0, 256,
             typesel, 256*256, 256, EP_BIAS);
        hgt_attn_kernel<<<(TOKL*NHEADS*LFR)/256, 256, 0, stream>>>(
            qkv, LIN(ly,10), LIN(ly,11), mask, meta, Bh2);
        convw(LIN(ly,8), Wh, 256, 256, 2);   // a_w
        gemm(Bh2, Wh, LIN(ly,9), feat, nullptr, 256, feat, 256, 256,
             typesel, 256*256, 256, EP_BIAS | EP_RESID);

        // ---------- Pyramid windows ----------
        ln(feat, LIN(ly,12), LIN(ly,13), Ah);
        for (int lvl = 0; lvl < 3; ++lvl) {
            const float* qkv_w = LIN(ly, 14 + 4*lvl + 0);
            const float* pos   = LIN(ly, 14 + 4*lvl + 1);
            const float* out_w = LIN(ly, 14 + 4*lvl + 2);
            const float* out_b = LIN(ly, 14 + 4*lvl + 3);
            convw(qkv_w, Wh, 256, 768, 1);
            gemm(Ah, Wh, nullptr, qkv, nullptr, 768, nullptr, 0, 768,
                 nullptr, 0, 0, 0);
            if (lvl == 0)
                pw_attn_kernel<4,16,16><<<(LFR*(HH/4)*(WW/4)*16*16)/256, 256, 0, stream>>>(qkv, pos, Bh2);
            else if (lvl == 1)
                pw_attn_kernel<8,8,32><<<(LFR*(HH/8)*(WW/8)*8*64)/256, 256, 0, stream>>>(qkv, pos, Bh2);
            else
                pw_attn_kernel<16,4,64><<<(LFR*(HH/16)*(WW/16)*4*256)/256, 256, 0, stream>>>(qkv, pos, Bh2);
            convw(out_w, Wh, 256, 256, 1);
            gemm(Bh2, Wh, out_b, pwacc, nullptr, 256, nullptr, 0, 256,
                 nullptr, 0, 0, EP_BIAS | (lvl ? EP_ACCUM : 0));
        }
        add_scale_kernel<<<(MTOT*DIMC)/256, 256, 0, stream>>>(feat, pwacc, 1.f/3.f);

        // ---------- FFN ----------
        ln(feat, LIN(ly,26), LIN(ly,27), Ah);
        convw(LIN(ly,28), Wh, 256, 256, 1);  // w1
        gemm(Ah, Wh, LIN(ly,29), nullptr, Bh2, 256, nullptr, 0, 256,
             nullptr, 0, 0, EP_BIAS | EP_GELU | EP_HALFOUT);
        convw(LIN(ly,30), Wh, 256, 256, 1);  // w2
        gemm(Bh2, Wh, LIN(ly,31), feat, nullptr, 256, feat, 256, 256,
             nullptr, 0, 0, EP_BIAS | EP_RESID);
    }

    (void)hipMemcpyAsync(d_out, feat, (size_t)MTOT*DIMC*sizeof(float),
                         hipMemcpyDeviceToDevice, stream);
    #undef LIN
}
